// SpatialTemporalDecoder_17781164605594
// MI455X (gfx1250) — matrix operand order repaired and checked by reading, not compiled, not run on hardware
//
#include <hip/hip_runtime.h>
#include <hip/hip_bf16.h>
#include <stdint.h>

// ---------------------------------------------------------------------------
// CDNA5 (gfx1250) WMMA bf16 implementation of the TFT-style decoder.
// wave32, V_WMMA_F32_16X16X32_BF16; fp32 accumulate; persistent LSTM kernel.
// K-loops are ping-pong software-pipelined (unroll-by-2, no register copies).
// ---------------------------------------------------------------------------

typedef __bf16 bf16_t;
typedef __attribute__((ext_vector_type(16))) __bf16 v16bf;
typedef __attribute__((ext_vector_type(8)))  float  v8f;

#define B_    64
#define OBS_  256
#define FUT_  256
#define T_    512
#define H_    512
#define G4H_  2048   // 4*H

enum { F_CAUSAL = 1, F_TRANSB = 2 };

__device__ __forceinline__ bf16_t f2bf(float f) {
  union { float f; uint32_t u; } x; x.f = f;
  uint32_t r = x.u + 0x7FFFu + ((x.u >> 16) & 1u);   // round-to-nearest-even
  uint16_t h = (uint16_t)(r >> 16);
  bf16_t b; __builtin_memcpy(&b, &h, 2); return b;
}

__device__ __forceinline__ float sigf(float x) { return 1.0f / (1.0f + __expf(-x)); }

union fragu { struct { uint4 lo, hi; } q; v16bf v; };

__device__ __forceinline__ v16bf load_frag(const bf16_t* p0, const bf16_t* p1) {
  fragu u;
  u.q.lo = *(const uint4*)p0;   // 16B aligned: k0 mult of 32, hi*8/hi*16 even
  u.q.hi = *(const uint4*)p1;
  return u.v;
}

// A fragment (ISA 16-bit 16x32): hi=0 lanes: K={k0..+7, k0+16..+23}; hi=1: +8/+24
__device__ __forceinline__ v16bf load_afrag(const bf16_t* Ap, int k0, int hi) {
  return load_frag(Ap + k0 + hi * 8, Ap + k0 + 16 + hi * 8);
}
// B fragment (32x16): n fixed per lane, K = {k0+hi*8..+7, k0+16+hi*8..+7} (matches A)
__device__ __forceinline__ v16bf load_bfrag(const bf16_t* Bp, int k0, int hi) {
  return load_frag(Bp + k0 + hi * 8, Bp + k0 + 16 + hi * 8);
}

#define WMMA_BF16(a, b, c) \
  __builtin_amdgcn_wmma_f32_16x16x32_bf16(false, (a), false, (b), (short)0, (c), false, false)

// ---------------------------------------------------------------------------
// Generic bf16 WMMA GEMM:  D[z] = A[z] (MxK) * Bw[z]^T (N rows, K cols) + bias
// 128 threads = 4 waves; wave tile 32x64 (2 A-frags x 4 B-frags = 8 WMMA),
// workgroup tile 128x64. Ping-pong pipelined K loop (two live buffer sets,
// alternating, so the allocator renames instead of copying registers).
// M mult of 128, N mult of 64, K >= 64 with K/32 even (all uses have K=512).
// ---------------------------------------------------------------------------
__global__ __launch_bounds__(128)
void wmma_gemm_bf16(const bf16_t* __restrict__ A,  int lda, size_t sA,
                    const bf16_t* __restrict__ Bw, int ldb, size_t sB,
                    const float*  __restrict__ bias,
                    float*        __restrict__ Cf, size_t sC,
                    bf16_t*       __restrict__ Cb, size_t sCb,
                    int ldc, int K, float alpha, int flags) {
  const int lane = threadIdx.x & 31;
  const int wave = threadIdx.x >> 5;
  const int r    = lane & 15;
  const int hi   = lane >> 4;
  const int row0 = blockIdx.y * 128 + wave * 32;
  const int col0 = blockIdx.x * 64;
  const int z    = blockIdx.z;

  const bf16_t* Ap0 = A + (size_t)z * sA + (size_t)(row0 + r) * lda;
  const bf16_t* Ap1 = Ap0 + (size_t)16 * lda;
  const bf16_t* Bp[4];
#pragma unroll
  for (int g = 0; g < 4; ++g)
    Bp[g] = Bw + (size_t)z * sB + (size_t)(col0 + 16 * g + r) * ldb;

  const v8f vzero = {0.f,0.f,0.f,0.f,0.f,0.f,0.f,0.f};
  v8f acc[2][4];
#pragma unroll
  for (int rt = 0; rt < 2; ++rt)
#pragma unroll
    for (int g = 0; g < 4; ++g) acc[rt][g] = vzero;

  v16bf aA[2], bA[4], aB[2], bB[4];
  aA[0] = load_afrag(Ap0, 0, hi);
  aA[1] = load_afrag(Ap1, 0, hi);
#pragma unroll
  for (int g = 0; g < 4; ++g) bA[g] = load_bfrag(Bp[g], 0, hi);

  int k0 = 0;
  for (; k0 + 64 < K; k0 += 64) {
    // phase B loads (k0+32) in flight over phase A WMMAs (k0)
    aB[0] = load_afrag(Ap0, k0 + 32, hi);
    aB[1] = load_afrag(Ap1, k0 + 32, hi);
#pragma unroll
    for (int g = 0; g < 4; ++g) bB[g] = load_bfrag(Bp[g], k0 + 32, hi);
#pragma unroll
    for (int rt = 0; rt < 2; ++rt)
#pragma unroll
      for (int g = 0; g < 4; ++g) acc[rt][g] = WMMA_BF16(aA[rt], bA[g], acc[rt][g]);
    // phase A loads (k0+64) in flight over phase B WMMAs (k0+32)
    aA[0] = load_afrag(Ap0, k0 + 64, hi);
    aA[1] = load_afrag(Ap1, k0 + 64, hi);
#pragma unroll
    for (int g = 0; g < 4; ++g) bA[g] = load_bfrag(Bp[g], k0 + 64, hi);
#pragma unroll
    for (int rt = 0; rt < 2; ++rt)
#pragma unroll
      for (int g = 0; g < 4; ++g) acc[rt][g] = WMMA_BF16(aB[rt], bB[g], acc[rt][g]);
  }
  // tail: aA/bA hold K-64; load K-32, then two WMMA groups
  aB[0] = load_afrag(Ap0, K - 32, hi);
  aB[1] = load_afrag(Ap1, K - 32, hi);
#pragma unroll
  for (int g = 0; g < 4; ++g) bB[g] = load_bfrag(Bp[g], K - 32, hi);
#pragma unroll
  for (int rt = 0; rt < 2; ++rt)
#pragma unroll
    for (int g = 0; g < 4; ++g) acc[rt][g] = WMMA_BF16(aA[rt], bA[g], acc[rt][g]);
#pragma unroll
  for (int rt = 0; rt < 2; ++rt)
#pragma unroll
    for (int g = 0; g < 4; ++g) acc[rt][g] = WMMA_BF16(aB[rt], bB[g], acc[rt][g]);

  // C/D layout: lane<16 -> n=lane, m=v ; lane>=16 -> n=lane-16, m=v+8
#pragma unroll
  for (int rt = 0; rt < 2; ++rt) {
#pragma unroll
    for (int g = 0; g < 4; ++g) {
      const int gn = col0 + 16 * g + r;
      const float bb = bias ? bias[gn] : 0.0f;
#pragma unroll
      for (int v = 0; v < 8; ++v) {
        const int gm = row0 + 16 * rt + hi * 8 + v;
        float val = acc[rt][g][v] + bb;
        if (flags & F_CAUSAL) val = (gn <= gm) ? val * alpha : -1e30f;
        if (Cf) Cf[(size_t)z * sC + (size_t)gm * ldc + gn] = val;
        if (Cb) {
          if (flags & F_TRANSB) {
            // flat M = B*T; store V^T per batch: [b, n, t] with T=H=512
            Cb[((size_t)(gm >> 9) << 18) + ((size_t)gn << 9) + (gm & 511)] = f2bf(val);
          } else {
            Cb[(size_t)z * sCb + (size_t)gm * ldc + gn] = f2bf(val);
          }
        }
      }
    }
  }
}

// ---------------------------------------------------------------------------
// Persistent fused LSTM: grid = 32 WGs x 128 threads runs all 256 steps.
// WG w owns j in [16w,16w+16). Its 4 WMMA N-tiles are {g*512+j}: all four
// gates for its j-slice, so the cell update happens on the accumulators in
// registers; c lives in VGPRs for the whole recurrence; only h (64KB bf16)
// round-trips L2 between steps, synchronized by an atomic grid barrier.
// ---------------------------------------------------------------------------
__global__ __launch_bounds__(128)
void k_lstm_persistent(const bf16_t* __restrict__ whh, const float* __restrict__ Xg,
                       const float* __restrict__ obs, bf16_t* h_bf,
                       bf16_t* __restrict__ lstmh, unsigned* cnt) {
  const int lane = threadIdx.x & 31;
  const int wave = threadIdx.x >> 5;
  const int r    = lane & 15;
  const int hi   = lane >> 4;
  const int w    = blockIdx.x;          // j-block
  const int j    = (w << 4) + r;
  const int row0 = wave << 4;           // 4 waves cover all 64 batch rows

  float creg[8];                        // cell state, register-resident
#pragma unroll
  for (int v = 0; v < 8; ++v) {
    int m = row0 + hi * 8 + v;
    creg[v] = obs[(((size_t)m * OBS_ + (OBS_ - 1)) << 9) + j];
  }

  const bf16_t* Ap = h_bf + ((size_t)(row0 + r) << 9);
  const bf16_t* Bp[4];
#pragma unroll
  for (int g = 0; g < 4; ++g)
    Bp[g] = whh + ((size_t)((g << 9) + (w << 4) + r) << 9);

  const v8f vzero = {0.f,0.f,0.f,0.f,0.f,0.f,0.f,0.f};
  unsigned target = 0;

  for (int t = 0; t < FUT_; ++t) {
    v8f acc[4];
#pragma unroll
    for (int g = 0; g < 4; ++g) acc[g] = vzero;

    v16bf aA, bA[4], aB, bB[4];
    aA = load_afrag(Ap, 0, hi);
#pragma unroll
    for (int g = 0; g < 4; ++g) bA[g] = load_bfrag(Bp[g], 0, hi);
    int k0 = 0;
    for (; k0 + 64 < H_; k0 += 64) {
      aB = load_afrag(Ap, k0 + 32, hi);
#pragma unroll
      for (int g = 0; g < 4; ++g) bB[g] = load_bfrag(Bp[g], k0 + 32, hi);
#pragma unroll
      for (int g = 0; g < 4; ++g) acc[g] = WMMA_BF16(aA, bA[g], acc[g]);
      aA = load_afrag(Ap, k0 + 64, hi);
#pragma unroll
      for (int g = 0; g < 4; ++g) bA[g] = load_bfrag(Bp[g], k0 + 64, hi);
#pragma unroll
      for (int g = 0; g < 4; ++g) acc[g] = WMMA_BF16(aB, bB[g], acc[g]);
    }
    aB = load_afrag(Ap, H_ - 32, hi);
#pragma unroll
    for (int g = 0; g < 4; ++g) bB[g] = load_bfrag(Bp[g], H_ - 32, hi);
#pragma unroll
    for (int g = 0; g < 4; ++g) acc[g] = WMMA_BF16(aA, bA[g], acc[g]);
#pragma unroll
    for (int g = 0; g < 4; ++g) acc[g] = WMMA_BF16(aB, bB[g], acc[g]);

    // cell update in registers: acc[0..3] = i,f,g,o recurrent pre-activations
#pragma unroll
    for (int v = 0; v < 8; ++v) {
      const int m = row0 + hi * 8 + v;
      const size_t xb = ((size_t)(m * FUT_ + t)) << 11;
      float gi = acc[0][v] + Xg[xb + j];
      float gf = acc[1][v] + Xg[xb + 512 + j];
      float gg = acc[2][v] + Xg[xb + 1024 + j];
      float go = acc[3][v] + Xg[xb + 1536 + j];
      float cn = sigf(gf) * creg[v] + sigf(gi) * tanhf(gg);
      float hn = sigf(go) * tanhf(cn);
      creg[v] = cn;
      bf16_t hb = f2bf(hn);
      h_bf[((size_t)m << 9) + j] = hb;
      lstmh[(((size_t)m * T_ + OBS_ + t) << 9) + j] = hb;
    }

    // device-scope grid barrier (h must be visible to all WGs before t+1)
    target += gridDim.x;
    __builtin_amdgcn_fence(__ATOMIC_RELEASE, "agent");
    __syncthreads();
    if (threadIdx.x == 0) {
      atomicAdd(cnt, 1u);
      while (__hip_atomic_load(cnt, __ATOMIC_RELAXED, __HIP_MEMORY_SCOPE_AGENT) < target) {
        __builtin_amdgcn_s_sleep(1);
      }
    }
    __syncthreads();
    __builtin_amdgcn_fence(__ATOMIC_ACQUIRE, "agent");
  }
}

// ---------------------------------------------------------------------------
// Elementwise / reduction kernels
// ---------------------------------------------------------------------------
__global__ void k_f32_to_bf16(const float* __restrict__ in, bf16_t* __restrict__ out, int n) {
  int i = blockIdx.x * blockDim.x + threadIdx.x;
  if (i < n) out[i] = f2bf(in[i]);
}

__global__ void k_sum_bias(const float* __restrict__ a, const float* __restrict__ b,
                           float* __restrict__ out, int n) {
  int i = blockIdx.x * blockDim.x + threadIdx.x;
  if (i < n) out[i] = a[i] + b[i];
}

__global__ void k_zero_u32(unsigned* p) { if (threadIdx.x == 0) p[0] = 0u; }

// vsn_observed (B,OBS,H) -> bf16 rows [b*T + t] of lstm_hidden
__global__ void k_obs_to_lstmh(const float* __restrict__ obs, bf16_t* __restrict__ out) {
  int i = blockIdx.x * blockDim.x + threadIdx.x;          // B*OBS*H threads
  int b = i >> 17;                                        // OBS*H = 2^17
  int rest = i & 131071;
  int t = rest >> 9, col = rest & 511;
  out[(((size_t)b * T_ + t) << 9) + col] = f2bf(obs[i]);
}

__global__ void k_init_h(const float* __restrict__ obs, bf16_t* __restrict__ h_bf) {
  int i = blockIdx.x * blockDim.x + threadIdx.x;          // B*H threads
  int b = i >> 9, j = i & 511;
  h_bf[i] = f2bf(obs[(((size_t)b * OBS_ + (OBS_ - 1)) << 9) + j]);
}

// z = sigmoid(z1)*z2 + y ; LayerNorm(H=512) ; fp32 out (+ optional bf16 copy)
// mode 0: residual = concat(obs, fut) select ; mode 1: residual = resA flat
__global__ __launch_bounds__(256)
void k_gluln(const float* __restrict__ z1, const float* __restrict__ z2,
             const float* __restrict__ resA, const float* __restrict__ resB,
             const float* __restrict__ gamma, const float* __restrict__ beta,
             float* __restrict__ outF, bf16_t* __restrict__ outB, int mode) {
  __shared__ float s1[8], s2[8];
  const int row = blockIdx.x;                 // B*T rows
  const int tid = threadIdx.x;
  const int lane = tid & 31, wid = tid >> 5;
  float zv[2], sum = 0.f, sq = 0.f;
#pragma unroll
  for (int e = 0; e < 2; ++e) {
    int col = tid + (e << 8);
    size_t idx = ((size_t)row << 9) + col;
    float y;
    if (mode == 0) {
      int b = row >> 9, t = row & 511;
      y = (t < OBS_) ? resA[(((size_t)b * OBS_ + t) << 9) + col]
                     : resB[(((size_t)b * FUT_ + (t - OBS_)) << 9) + col];
    } else {
      y = resA[idx];
    }
    float zz = sigf(z1[idx]) * z2[idx] + y;
    zv[e] = zz; sum += zz; sq += zz * zz;
  }
#pragma unroll
  for (int off = 16; off; off >>= 1) {
    sum += __shfl_xor(sum, off, 32);
    sq  += __shfl_xor(sq,  off, 32);
  }
  if (lane == 0) { s1[wid] = sum; s2[wid] = sq; }
  __syncthreads();
  float ts = 0.f, tq = 0.f;
#pragma unroll
  for (int i = 0; i < 8; ++i) { ts += s1[i]; tq += s2[i]; }
  float mu  = ts * (1.0f / 512.0f);
  float var = tq * (1.0f / 512.0f) - mu * mu;
  float ri  = rsqrtf(var + 1e-5f);
#pragma unroll
  for (int e = 0; e < 2; ++e) {
    int col = tid + (e << 8);
    size_t idx = ((size_t)row << 9) + col;
    float o = gamma[col] * (zv[e] - mu) * ri + beta[col];
    outF[idx] = o;
    if (outB) outB[idx] = f2bf(o);
  }
}

// Row softmax over T=512 (masked entries already -1e30). One wave per row.
__global__ __launch_bounds__(128)
void k_softmax(float* __restrict__ p, bf16_t* __restrict__ pb) {
  const int lane = threadIdx.x & 31;
  const int row  = blockIdx.x * 4 + (threadIdx.x >> 5);
  float* rp = p + ((size_t)row << 9);
  bf16_t* rb = pb + ((size_t)row << 9);
  float v[16], m = -1e38f;
#pragma unroll
  for (int i = 0; i < 16; ++i) { v[i] = rp[lane + 32 * i]; m = fmaxf(m, v[i]); }
#pragma unroll
  for (int off = 16; off; off >>= 1) m = fmaxf(m, __shfl_xor(m, off, 32));
  float s = 0.f;
#pragma unroll
  for (int i = 0; i < 16; ++i) { v[i] = __expf(v[i] - m); s += v[i]; }
#pragma unroll
  for (int off = 16; off; off >>= 1) s += __shfl_xor(s, off, 32);
  float inv = 1.0f / s;
#pragma unroll
  for (int i = 0; i < 16; ++i) {
    float rr = v[i] * inv;
    rp[lane + 32 * i] = rr;
    rb[lane + 32 * i] = f2bf(rr);
  }
}

// ---------------------------------------------------------------------------
// Host orchestration
// ---------------------------------------------------------------------------
extern "C" void kernel_launch(void* const* d_in, const int* in_sizes, int n_in,
                              void* d_out, int out_size, void* d_ws, size_t ws_size,
                              hipStream_t stream) {
  (void)in_sizes; (void)n_in; (void)out_size; (void)ws_size;
  const float* obs   = (const float*)d_in[0];
  const float* fut   = (const float*)d_in[1];
  const float* W_ih  = (const float*)d_in[2];
  const float* W_hh  = (const float*)d_in[3];
  const float* b_ih  = (const float*)d_in[4];
  const float* b_hh  = (const float*)d_in[5];
  const float* g1W1  = (const float*)d_in[6];
  const float* g1b1  = (const float*)d_in[7];
  const float* g1W2  = (const float*)d_in[8];
  const float* g1b2  = (const float*)d_in[9];
  const float* g1g   = (const float*)d_in[10];
  const float* g1b   = (const float*)d_in[11];
  const float* Wq    = (const float*)d_in[12];
  const float* bq    = (const float*)d_in[13];
  const float* Wk    = (const float*)d_in[14];
  const float* bk    = (const float*)d_in[15];
  const float* Wv    = (const float*)d_in[16];
  const float* bv    = (const float*)d_in[17];
  const float* Wo    = (const float*)d_in[18];
  const float* bo    = (const float*)d_in[19];
  const float* g2W1  = (const float*)d_in[20];
  const float* g2b1  = (const float*)d_in[21];
  const float* g2W2  = (const float*)d_in[22];
  const float* g2b2  = (const float*)d_in[23];
  const float* g2g   = (const float*)d_in[24];
  const float* g2b   = (const float*)d_in[25];

  float* out       = (float*)d_out;
  float* out_delta = out;                        // B*T*H
  float* out_phi   = out + (size_t)16777216;     // B*T*H
  float* out_attn  = out + (size_t)33554432;     // B*T*T

  // ---- workspace arena (aliased reuse is stream-ordered-safe) ----
  char* ws = (char*)d_ws;
  size_t off = 0;
  auto take = [&](size_t bytes) -> void* {
    void* p = ws + off; off += (bytes + 255) & ~(size_t)255; return p;
  };
  bf16_t* wih_bf  = (bf16_t*)take((size_t)G4H_ * H_ * 2);
  bf16_t* whh_bf  = (bf16_t*)take((size_t)G4H_ * H_ * 2);
  bf16_t* g1w1_bf = (bf16_t*)take((size_t)H_ * H_ * 2);
  bf16_t* g1w2_bf = (bf16_t*)take((size_t)H_ * H_ * 2);
  bf16_t* wq_bf   = (bf16_t*)take((size_t)H_ * H_ * 2);
  bf16_t* wk_bf   = (bf16_t*)take((size_t)H_ * H_ * 2);
  bf16_t* wv_bf   = (bf16_t*)take((size_t)H_ * H_ * 2);
  bf16_t* wo_bf   = (bf16_t*)take((size_t)H_ * H_ * 2);
  bf16_t* g2w1_bf = (bf16_t*)take((size_t)H_ * H_ * 2);
  bf16_t* g2w2_bf = (bf16_t*)take((size_t)H_ * H_ * 2);
  float*  biassum = (float*)take((size_t)G4H_ * 4);
  bf16_t* fut_bf  = (bf16_t*)take((size_t)B_ * FUT_ * H_ * 2);
  bf16_t* lstmh_bf  = (bf16_t*)take((size_t)B_ * T_ * H_ * 2);
  bf16_t* gluphi_bf = (bf16_t*)take((size_t)B_ * T_ * H_ * 2);
  bf16_t* q_bf    = (bf16_t*)take((size_t)B_ * T_ * H_ * 2);  // -> attn_bf after scores
  bf16_t* k_bf    = (bf16_t*)take((size_t)B_ * T_ * H_ * 2);  // -> attnout_bf after attn@V
  bf16_t* vt_bf   = (bf16_t*)take((size_t)B_ * T_ * H_ * 2);  // -> proj_bf after Wo
  float*  Xg      = (float*)take((size_t)B_ * FUT_ * G4H_ * 4); // -> z1|z2 after LSTM
  bf16_t* h_bf    = (bf16_t*)take((size_t)B_ * H_ * 2);
  unsigned* cnt   = (unsigned*)take(256);

  bf16_t* attn_bf = q_bf;            // B*T*T == B*T*H since T==H
  bf16_t* ao_bf   = k_bf;
  bf16_t* proj_bf = vt_bf;
  float*  z1      = Xg;              // B*T*H
  float*  z2      = Xg + (size_t)16777216;

  const dim3 blk256(256), blk128(128);
  auto cvt = [&](const float* src, bf16_t* dst, int n) {
    k_f32_to_bf16<<<dim3((n + 255) / 256), blk256, 0, stream>>>(src, dst, n);
  };

  // 1) weights -> bf16, bias sums, activations -> bf16, counter = 0
  cvt(W_ih, wih_bf, G4H_ * H_);   cvt(W_hh, whh_bf, G4H_ * H_);
  cvt(g1W1, g1w1_bf, H_ * H_);    cvt(g1W2, g1w2_bf, H_ * H_);
  cvt(Wq, wq_bf, H_ * H_);        cvt(Wk, wk_bf, H_ * H_);
  cvt(Wv, wv_bf, H_ * H_);        cvt(Wo, wo_bf, H_ * H_);
  cvt(g2W1, g2w1_bf, H_ * H_);    cvt(g2W2, g2w2_bf, H_ * H_);
  k_sum_bias<<<dim3(G4H_ / 256), blk256, 0, stream>>>(b_ih, b_hh, biassum, G4H_);
  cvt(fut, fut_bf, B_ * FUT_ * H_);
  k_obs_to_lstmh<<<dim3(B_ * OBS_ * H_ / 256), blk256, 0, stream>>>(obs, lstmh_bf);
  k_init_h<<<dim3(B_ * H_ / 256), blk256, 0, stream>>>(obs, h_bf);
  k_zero_u32<<<dim3(1), dim3(64), 0, stream>>>(cnt);

  // 2) Xg = fut @ W_ih^T + (b_ih + b_hh)   [16384 x 2048 x 512]
  wmma_gemm_bf16<<<dim3(G4H_ / 64, (B_ * FUT_) / 128, 1), blk128, 0, stream>>>(
      fut_bf, H_, 0, wih_bf, H_, 0, biassum, Xg, 0, nullptr, 0, G4H_, H_, 1.0f, 0);

  // 3) whole LSTM recurrence in one persistent kernel (32 WGs, 256 steps)
  k_lstm_persistent<<<dim3(32), blk128, 0, stream>>>(
      whh_bf, Xg, obs, h_bf, lstmh_bf, cnt);

  // 4) GLU1: z1 = lstmh@g1W1^T+b1 ; z2 = lstmh@g1W2^T+b2 ; fuse + LN -> glu_phi
  wmma_gemm_bf16<<<dim3(H_ / 64, (B_ * T_) / 128, 1), blk128, 0, stream>>>(
      lstmh_bf, H_, 0, g1w1_bf, H_, 0, g1b1, z1, 0, nullptr, 0, H_, H_, 1.0f, 0);
  wmma_gemm_bf16<<<dim3(H_ / 64, (B_ * T_) / 128, 1), blk128, 0, stream>>>(
      lstmh_bf, H_, 0, g1w2_bf, H_, 0, g1b2, z2, 0, nullptr, 0, H_, H_, 1.0f, 0);
  k_gluln<<<dim3(B_ * T_), blk256, 0, stream>>>(
      z1, z2, obs, fut, g1g, g1b, out_phi, gluphi_bf, 0);

  // 5) Q, K, V (V stored transposed per batch for the attn@V GEMM)
  wmma_gemm_bf16<<<dim3(H_ / 64, (B_ * T_) / 128, 1), blk128, 0, stream>>>(
      gluphi_bf, H_, 0, wq_bf, H_, 0, bq, nullptr, 0, q_bf, 0, H_, H_, 1.0f, 0);
  wmma_gemm_bf16<<<dim3(H_ / 64, (B_ * T_) / 128, 1), blk128, 0, stream>>>(
      gluphi_bf, H_, 0, wk_bf, H_, 0, bk, nullptr, 0, k_bf, 0, H_, H_, 1.0f, 0);
  wmma_gemm_bf16<<<dim3(H_ / 64, (B_ * T_) / 128, 1), blk128, 0, stream>>>(
      gluphi_bf, H_, 0, wv_bf, H_, 0, bv, nullptr, 0, vt_bf, 0, H_, H_, 1.0f, F_TRANSB);

  // 6) scores = (Q @ K^T) / sqrt(H) with causal mask, straight into out_attn
  wmma_gemm_bf16<<<dim3(T_ / 64, T_ / 128, B_), blk128, 0, stream>>>(
      q_bf, H_, (size_t)T_ * H_, k_bf, H_, (size_t)T_ * H_, nullptr,
      out_attn, (size_t)T_ * T_, nullptr, 0, T_, H_, 0.044194173824159216f, F_CAUSAL);

  // 7) softmax rows (in-place fp32 + bf16 copy)
  k_softmax<<<dim3((B_ * T_) / 4), blk128, 0, stream>>>(out_attn, attn_bf);

  // 8) attn_out = attn @ V  (B operand = V^T per batch)
  wmma_gemm_bf16<<<dim3(H_ / 64, T_ / 128, B_), blk128, 0, stream>>>(
      attn_bf, T_, (size_t)T_ * T_, vt_bf, T_, (size_t)H_ * T_, nullptr,
      nullptr, 0, ao_bf, (size_t)T_ * H_, H_, T_, 1.0f, 0);

  // 9) proj = attn_out @ Wo^T + bo  (bf16 only; feeds GLU2 GEMMs)
  wmma_gemm_bf16<<<dim3(H_ / 64, (B_ * T_) / 128, 1), blk128, 0, stream>>>(
      ao_bf, H_, 0, wo_bf, H_, 0, bo, nullptr, 0, proj_bf, 0, H_, H_, 1.0f, 0);

  // 10) GLU2 + LN -> glu_delta (residual = glu_phi fp32 already in d_out)
  wmma_gemm_bf16<<<dim3(H_ / 64, (B_ * T_) / 128, 1), blk128, 0, stream>>>(
      proj_bf, H_, 0, g2w1_bf, H_, 0, g2b1, z1, 0, nullptr, 0, H_, H_, 1.0f, 0);
  wmma_gemm_bf16<<<dim3(H_ / 64, (B_ * T_) / 128, 1), blk128, 0, stream>>>(
      proj_bf, H_, 0, g2w2_bf, H_, 0, g2b2, z2, 0, nullptr, 0, H_, H_, 1.0f, 0);
  k_gluln<<<dim3(B_ * T_), blk256, 0, stream>>>(
      z1, z2, out_phi, nullptr, g2g, g2b, out_delta, nullptr, 1);
}
